// PointNetV2FeatureLearnerMultiScaleGrouping_16157666967953
// MI455X (gfx1250) — matrix operand order repaired and checked by reading, not compiled, not run on hardware
//
#include <hip/hip_runtime.h>

typedef __attribute__((ext_vector_type(16))) _Float16 v16h;
typedef __attribute__((ext_vector_type(8)))  _Float16 v8h;
typedef __attribute__((ext_vector_type(8)))  float    v8f;

__host__ __device__ constexpr int imax(int a, int b) { return a > b ? a : b; }
__host__ __device__ constexpr int pad32(int c) { return (c + 31) & ~31; }

#define BN_INV 0.9999950000374997f   // 1/sqrt(1+1e-5)

// ---------------------------------------------------------------------------
// (B,3,N) -> (B,N,3)
__global__ void transpose_xyz_kernel(const float* __restrict__ x, float* __restrict__ xyz) {
    constexpr int B = 8, N = 4096;
    int t = blockIdx.x * blockDim.x + threadIdx.x;
    if (t >= B * N * 3) return;
    int c = t % 3;
    int nb = t / 3;
    int n = nb % N;
    int b = nb / N;
    xyz[t] = x[((size_t)b * 3 + c) * N + n];
}

// W (cout,cin) f32 -> Wh (cout,cinpad) f16 zero-padded
__global__ void convert_w_kernel(const float* __restrict__ W, _Float16* __restrict__ wh,
                                 int cout, int cin, int cinpad) {
    int t = blockIdx.x * blockDim.x + threadIdx.x;
    if (t >= cout * cinpad) return;
    int o = t / cinpad, k = t % cinpad;
    wh[t] = (k < cin) ? (_Float16)W[o * cin + k] : (_Float16)0.0f;
}

// ---------------------------------------------------------------------------
// Farthest point sampling: one block per batch, points+distance live in LDS.
template<int N, int NPOINT>
__global__ void __launch_bounds__(256) fps_kernel(const float* __restrict__ xyz,
                                                  float* __restrict__ new_xyz) {
    extern __shared__ float fsm[];
    float* px = fsm;
    float* py = px + N;
    float* pz = py + N;
    float* dist = pz + N;
    float* rval = dist + N;
    int*   ridx = (int*)(rval + 256);
    __shared__ int s_far;

    int b = blockIdx.x, tid = threadIdx.x;
    const float* base = xyz + (size_t)b * N * 3;
    for (int j = tid; j < N; j += 256) {
        px[j] = base[j * 3 + 0];
        py[j] = base[j * 3 + 1];
        pz[j] = base[j * 3 + 2];
        dist[j] = 1e10f;
    }
    if (tid == 0) s_far = 0;
    __syncthreads();

    for (int i = 0; i < NPOINT; i++) {
        int far = s_far;
        float cx = px[far], cy = py[far], cz = pz[far];
        if (tid == 0) {
            float* o = new_xyz + ((size_t)b * NPOINT + i) * 3;
            o[0] = cx; o[1] = cy; o[2] = cz;
        }
        float best = -1.0f; int bidx = 0;
        for (int j = tid; j < N; j += 256) {
            float dx = px[j] - cx, dy = py[j] - cy, dz = pz[j] - cz;
            float d = dx * dx + dy * dy + dz * dz;
            float dj = fminf(dist[j], d);
            dist[j] = dj;
            if (dj > best) { best = dj; bidx = j; }
        }
        rval[tid] = best; ridx[tid] = bidx;
        __syncthreads();
        for (int s = 128; s > 0; s >>= 1) {
            if (tid < s) {
                float ov = rval[tid + s]; int oi = ridx[tid + s];
                if (ov > rval[tid] || (ov == rval[tid] && oi < ridx[tid])) {
                    rval[tid] = ov; ridx[tid] = oi;
                }
            }
            __syncthreads();
        }
        if (tid == 0) s_far = ridx[0];
        __syncthreads();
    }
}

// ---------------------------------------------------------------------------
// One MLP layer in LDS via v_wmma_f32_16x16x32_f16, one 16x16 tile per wave.
template<int K, int INW, int COUT>
__device__ __forceinline__ void mlp_layer_lds(const _Float16* __restrict__ bufIn,
                                              _Float16* __restrict__ bufOut,
                                              const _Float16* __restrict__ wh,
                                              const float* __restrict__ bb,
                                              const float* __restrict__ gg,
                                              const float* __restrict__ ee,
                                              int wave, int lane) {
    constexpr int COLT = COUT / 16;
    constexpr int TILES = (K / 16) * COLT;
    const int lg  = lane >> 4;
    const int l15 = lane & 15;
    for (int t = wave; t < TILES; t += 4) {
        int rt = t / COLT, ct = t % COLT;
        v8f acc = {};
        int rowA = rt * 16 + l15;
        int colB = ct * 16 + l15;
        const _Float16* aBase = bufIn + rowA * INW + lg * 16;
        const _Float16* bBase = wh + (size_t)colB * INW + lg * 16;
        for (int kk = 0; kk < INW; kk += 32) {
            v8h lo = *(const v8h*)(aBase + kk);        // A: K offsets lg*16 .. lg*16+7
            v8h hi = *(const v8h*)(aBase + kk + 8);    // A: +8
            v16h a = __builtin_shufflevector(lo, hi, 0,1,2,3,4,5,6,7,8,9,10,11,12,13,14,15);
            v16h bm = *(const v16h*)(bBase + kk);      // B: K = lg*16 + e, contiguous
            acc = __builtin_amdgcn_wmma_f32_16x16x32_f16(false, a, false, bm,
                                                         (short)0, acc, false, false);
        }
        int col = ct * 16 + l15;
        float G = gg[col], Bv = bb[col], E = ee[col];
#pragma unroll
        for (int r = 0; r < 8; r++) {
            int row = rt * 16 + r + lg * 8;
            float y = G * ((acc[r] + Bv) * BN_INV) + E;
            y = y > 0.0f ? y : 0.0f;
            bufOut[row * COUT + col] = (_Float16)y;
        }
    }
}

// ---------------------------------------------------------------------------
// Fused ball-query + gather + 3-layer MLP + max-pool. One block per (b, centroid).
template<int NSRC, int M, int K, int CINPAD, int C1, int C2, int C3, bool HASPTS, int CPTS>
__global__ void __launch_bounds__(128)
branch_kernel(const float* __restrict__ xyz,          // (B,NSRC,3)
              const float* __restrict__ pts,          // (B,NSRC,CPTS) or null
              const float* __restrict__ nxyz,         // (B,M,3)
              float r2,
              const _Float16* __restrict__ wh1, const float* __restrict__ bb1,
              const float* __restrict__ gg1, const float* __restrict__ ee1,
              const _Float16* __restrict__ wh2, const float* __restrict__ bb2,
              const float* __restrict__ gg2, const float* __restrict__ ee2,
              const _Float16* __restrict__ wh3, const float* __restrict__ bb3,
              const float* __restrict__ gg3, const float* __restrict__ ee3,
              float* __restrict__ outFeat, int outStride, int chOff) {
    constexpr int BUFAW = imax(CINPAD, C2);
    constexpr int BUFBW = imax(C1, C3);
    extern __shared__ _Float16 smemh[];
    _Float16* bufA = smemh;                  // K*BUFAW halves
    _Float16* bufB = bufA + K * BUFAW;       // K*BUFBW halves
    int*      sidx = (int*)(bufB + K * BUFBW);
    __shared__ int scount;

    const int tid  = threadIdx.x;
    const int lane = tid & 31;
    const int wave = tid >> 5;
    const int b = blockIdx.x / M;
    const int m = blockIdx.x % M;
    const float* nc = nxyz + ((size_t)b * M + m) * 3;

    // ---- ball query: first K in-radius indices, ascending (wave0 ballot scan)
    if (wave == 0) {
        float cx = nc[0], cy = nc[1], cz = nc[2];
        int count = 0;
        for (int base = 0; base < NSRC && count < K; base += 32) {
            int j = base + lane;
            const float* p = xyz + ((size_t)b * NSRC + j) * 3;
            float dx = p[0] - cx, dy = p[1] - cy, dz = p[2] - cz;
            float d2 = dx * dx + dy * dy + dz * dz;
            bool in = (d2 <= r2);
            unsigned mask = __builtin_amdgcn_ballot_w32(in);
            int pre = __builtin_popcount(mask & ((1u << lane) - 1u));
            int pos = count + pre;
            if (in && pos < K) sidx[pos] = j;
            count += __builtin_popcount(mask);
        }
        if (count > K) count = K;
        if (lane == 0) scount = count;
    }
    __syncthreads();
    int cnt = scount;
    for (int p = cnt + tid; p < K; p += 128) sidx[p] = sidx[0];
    __syncthreads();

    // ---- gather grouped features into LDS as f16 (zero-padded channels)
    for (int t = tid; t < K * CINPAD; t += 128) {
        int r = t / CINPAD, c = t % CINPAD;
        int src = sidx[r];
        float v = 0.0f;
        if constexpr (HASPTS) {
            if (c < CPTS)             v = pts[((size_t)b * NSRC + src) * CPTS + c];
            else if (c < CPTS + 3)    v = xyz[((size_t)b * NSRC + src) * 3 + (c - CPTS)] - nc[c - CPTS];
        } else {
            if (c < 3)                v = xyz[((size_t)b * NSRC + src) * 3 + c] - nc[c];
        }
        bufA[t] = (_Float16)v;
    }
    __syncthreads();

    // ---- 3 WMMA MLP layers, ping-pong in LDS
    mlp_layer_lds<K, CINPAD, C1>(bufA, bufB, wh1, bb1, gg1, ee1, wave, lane);
    __syncthreads();
    mlp_layer_lds<K, C1, C2>(bufB, bufA, wh2, bb2, gg2, ee2, wave, lane);
    __syncthreads();
    mlp_layer_lds<K, C2, C3>(bufA, bufB, wh3, bb3, gg3, ee3, wave, lane);
    __syncthreads();

    // ---- max-pool over K samples
    for (int c = tid; c < C3; c += 128) {
        float mx = -1e30f;
        for (int r = 0; r < K; r++) mx = fmaxf(mx, (float)bufB[r * C3 + c]);
        outFeat[((size_t)b * M + m) * outStride + chOff + c] = mx;
    }
}

// ---------------------------------------------------------------------------
// sa3: build (B,128,672) f16 input = concat(xyz, feat) zero-padded
__global__ void build_gp3_kernel(const float* __restrict__ nx2,
                                 const float* __restrict__ feat,
                                 _Float16* __restrict__ gph) {
    constexpr int B = 8, R = 128, CP = 672;
    int t = blockIdx.x * blockDim.x + threadIdx.x;
    if (t >= B * R * CP) return;
    int c = t % CP;
    int rb = t / CP;
    int r = rb % R;
    int b = rb / R;
    float v = 0.0f;
    if (c < 3)        v = nx2[((size_t)b * R + r) * 3 + c];
    else if (c < 643) v = feat[((size_t)b * R + r) * 640 + (c - 3)];
    gph[t] = (_Float16)v;
}

// sa3 layer: global-memory f16 activations, same WMMA tile body
template<int ROWS, int INW, int COUT>
__global__ void __launch_bounds__(128)
gemm_layer_kernel(const _Float16* __restrict__ Ain,
                  const _Float16* __restrict__ wh,
                  const float* __restrict__ bb, const float* __restrict__ gg,
                  const float* __restrict__ ee,
                  _Float16* __restrict__ Aout) {
    constexpr int COLT = COUT / 16;
    constexpr int TILES = (ROWS / 16) * COLT;
    const int b = blockIdx.x;
    const int wave = threadIdx.x >> 5;
    const int lane = threadIdx.x & 31;
    const int lg = lane >> 4, l15 = lane & 15;
    const _Float16* A = Ain + (size_t)b * ROWS * INW;
    _Float16* O = Aout + (size_t)b * ROWS * COUT;
    for (int t = blockIdx.y * 4 + wave; t < TILES; t += gridDim.y * 4) {
        int rt = t / COLT, ct = t % COLT;
        v8f acc = {};
        int rowA = rt * 16 + l15;
        int colB = ct * 16 + l15;
        const _Float16* aBase = A + (size_t)rowA * INW + lg * 16;
        const _Float16* bBase = wh + (size_t)colB * INW + lg * 16;
        for (int kk = 0; kk < INW; kk += 32) {
            v8h lo = *(const v8h*)(aBase + kk);
            v8h hi = *(const v8h*)(aBase + kk + 8);
            v16h a = __builtin_shufflevector(lo, hi, 0,1,2,3,4,5,6,7,8,9,10,11,12,13,14,15);
            v16h bm = *(const v16h*)(bBase + kk);
            acc = __builtin_amdgcn_wmma_f32_16x16x32_f16(false, a, false, bm,
                                                         (short)0, acc, false, false);
        }
        int col = ct * 16 + l15;
        float G = gg[col], Bv = bb[col], E = ee[col];
#pragma unroll
        for (int r = 0; r < 8; r++) {
            int row = rt * 16 + r + lg * 8;
            float y = G * ((acc[r] + Bv) * BN_INV) + E;
            y = y > 0.0f ? y : 0.0f;
            O[(size_t)row * COUT + col] = (_Float16)y;
        }
    }
}

// sa3 max over 128 rows -> write both output halves (x_flat and its transpose view)
__global__ void sa3_max_kernel(const _Float16* __restrict__ act, float* __restrict__ out) {
    constexpr int B = 8, R = 128, C = 1024;
    int b = blockIdx.x;
    for (int c = threadIdx.x; c < C; c += blockDim.x) {
        float mx = -1e30f;
        for (int r = 0; r < R; r++)
            mx = fmaxf(mx, (float)act[((size_t)b * R + r) * C + c]);
        out[b * C + c] = mx;
        out[B * C + b * C + c] = mx;
    }
}

// ---------------------------------------------------------------------------
struct LP { const float *W, *b, *g, *be; };

template<int NSRC, int M, int K, int CINPAD, int C1, int C2, int C3, bool HASPTS, int CPTS>
static void launch_branch(hipStream_t stream,
                          const float* xyz, const float* pts, const float* nxyz, float r2,
                          const _Float16* w1, const LP& L1,
                          const _Float16* w2, const LP& L2,
                          const _Float16* w3, const LP& L3,
                          float* outFeat, int outStride, int chOff) {
    constexpr int BUFAW = imax(CINPAD, C2);
    constexpr int BUFBW = imax(C1, C3);
    int shm = K * (BUFAW + BUFBW) * 2 + K * 4 + 32;
    auto kfn = branch_kernel<NSRC, M, K, CINPAD, C1, C2, C3, HASPTS, CPTS>;
    if (shm > 65536)
        hipFuncSetAttribute(reinterpret_cast<const void*>(kfn),
                            hipFuncAttributeMaxDynamicSharedMemorySize, shm);
    kfn<<<dim3(8 * M), 128, shm, stream>>>(xyz, pts, nxyz, r2,
        w1, L1.b, L1.g, L1.be, w2, L2.b, L2.g, L2.be, w3, L3.b, L3.g, L3.be,
        outFeat, outStride, chOff);
}

extern "C" void kernel_launch(void* const* d_in, const int* in_sizes, int n_in,
                              void* d_out, int out_size, void* d_ws, size_t ws_size,
                              hipStream_t stream) {
    constexpr int B = 8, N = 4096;
    const float* x = (const float*)d_in[0];

    // ---- collect layer params in flatten order: x, sa1[3][3], sa2[3][3], sa3[3]
    LP sa1[3][3], sa2[3][3], sa3[3];
    int p = 1;
    auto take = [&](LP& L) {
        L.W  = (const float*)d_in[p + 0];
        L.b  = (const float*)d_in[p + 1];
        L.g  = (const float*)d_in[p + 2];
        L.be = (const float*)d_in[p + 3];
        p += 4;
    };
    for (int br = 0; br < 3; br++) for (int l = 0; l < 3; l++) take(sa1[br][l]);
    for (int br = 0; br < 3; br++) for (int l = 0; l < 3; l++) take(sa2[br][l]);
    for (int l = 0; l < 3; l++) take(sa3[l]);

    static const int sa1_dims[3][4] = {{3,32,32,64},{3,64,64,128},{3,64,96,128}};
    static const int sa2_dims[3][4] = {{323,64,64,128},{323,128,128,256},{323,128,128,256}};
    static const int sa3_dims[4]    = {643,256,512,1024};

    // ---- carve workspace
    char* wsb = (char*)d_ws;
    size_t off = 0;
    auto alloc = [&](size_t bytes) -> char* {
        char* r = wsb + off;
        off = (off + bytes + 255) & ~(size_t)255;
        return r;
    };
    float* xyz = (float*)alloc((size_t)B * N * 3 * 4);
    float* nx1 = (float*)alloc((size_t)B * 512 * 3 * 4);
    float* nx2 = (float*)alloc((size_t)B * 128 * 3 * 4);
    float* f1  = (float*)alloc((size_t)B * 512 * 320 * 4);
    float* f2  = (float*)alloc((size_t)B * 128 * 640 * 4);
    _Float16* gph  = (_Float16*)alloc((size_t)B * 128 * 672 * 2);
    _Float16* actA = (_Float16*)alloc((size_t)B * 128 * 1024 * 2);
    _Float16* actB = (_Float16*)alloc((size_t)B * 128 * 1024 * 2);

    _Float16 *wh1[3][3], *wh2[3][3], *wh3[3];
    auto convW = [&](const LP& L, int cin, int cout) -> _Float16* {
        int cp = pad32(cin);
        _Float16* dst = (_Float16*)alloc((size_t)cout * cp * 2);
        int total = cout * cp;
        convert_w_kernel<<<(total + 255) / 256, 256, 0, stream>>>(L.W, dst, cout, cin, cp);
        return dst;
    };
    for (int br = 0; br < 3; br++)
        for (int l = 0; l < 3; l++)
            wh1[br][l] = convW(sa1[br][l], sa1_dims[br][l], sa1_dims[br][l + 1]);
    for (int br = 0; br < 3; br++)
        for (int l = 0; l < 3; l++)
            wh2[br][l] = convW(sa2[br][l], sa2_dims[br][l], sa2_dims[br][l + 1]);
    for (int l = 0; l < 3; l++)
        wh3[l] = convW(sa3[l], sa3_dims[l], sa3_dims[l + 1]);

    // ---- stage 0: transpose
    {
        int total = B * N * 3;
        transpose_xyz_kernel<<<(total + 255) / 256, 256, 0, stream>>>(x, xyz);
    }

    // ---- sa1: FPS(4096 -> 512)
    {
        int shm = (4 * N + 512) * 4;
        hipFuncSetAttribute(reinterpret_cast<const void*>(fps_kernel<N, 512>),
                            hipFuncAttributeMaxDynamicSharedMemorySize, shm);
        fps_kernel<N, 512><<<B, 256, shm, stream>>>(xyz, nx1);
    }
    // sa1 branches: radii 0.1/0.2/0.4, K 16/32/128
    launch_branch<4096, 512, 16, 32, 32, 32, 64, false, 0>(stream, xyz, nullptr, nx1, 0.01f,
        wh1[0][0], sa1[0][0], wh1[0][1], sa1[0][1], wh1[0][2], sa1[0][2], f1, 320, 0);
    launch_branch<4096, 512, 32, 32, 64, 64, 128, false, 0>(stream, xyz, nullptr, nx1, 0.04f,
        wh1[1][0], sa1[1][0], wh1[1][1], sa1[1][1], wh1[1][2], sa1[1][2], f1, 320, 64);
    launch_branch<4096, 512, 128, 32, 64, 96, 128, false, 0>(stream, xyz, nullptr, nx1, 0.16f,
        wh1[2][0], sa1[2][0], wh1[2][1], sa1[2][1], wh1[2][2], sa1[2][2], f1, 320, 192);

    // ---- sa2: FPS(512 -> 128)
    {
        int shm = (4 * 512 + 512) * 4;
        fps_kernel<512, 128><<<B, 256, shm, stream>>>(nx1, nx2);
    }
    // sa2 branches: radii 0.2/0.4/0.8, K 32/64/128, pts = sa1_feat (320ch)
    launch_branch<512, 128, 32, 352, 64, 64, 128, true, 320>(stream, nx1, f1, nx2, 0.04f,
        wh2[0][0], sa2[0][0], wh2[0][1], sa2[0][1], wh2[0][2], sa2[0][2], f2, 640, 0);
    launch_branch<512, 128, 64, 352, 128, 128, 256, true, 320>(stream, nx1, f1, nx2, 0.16f,
        wh2[1][0], sa2[1][0], wh2[1][1], sa2[1][1], wh2[1][2], sa2[1][2], f2, 640, 128);
    launch_branch<512, 128, 128, 352, 128, 128, 256, true, 320>(stream, nx1, f1, nx2, 0.64f,
        wh2[2][0], sa2[2][0], wh2[2][1], sa2[2][1], wh2[2][2], sa2[2][2], f2, 640, 384);

    // ---- sa3: group-all MLP 643 -> 256 -> 512 -> 1024, max over 128
    {
        int total = B * 128 * 672;
        build_gp3_kernel<<<(total + 255) / 256, 256, 0, stream>>>(nx2, f2, gph);
    }
    gemm_layer_kernel<128, 672, 256><<<dim3(B, 32), 128, 0, stream>>>(
        gph, wh3[0], sa3[0].b, sa3[0].g, sa3[0].be, actA);
    gemm_layer_kernel<128, 256, 512><<<dim3(B, 64), 128, 0, stream>>>(
        actA, wh3[1], sa3[1].b, sa3[1].g, sa3[1].be, actB);
    gemm_layer_kernel<128, 512, 1024><<<dim3(B, 128), 128, 0, stream>>>(
        actB, wh3[2], sa3[2].b, sa3[2].g, sa3[2].be, actA);
    sa3_max_kernel<<<B, 256, 0, stream>>>(actA, (float*)d_out);
}
